// SimpleLSTM_3985729651282
// MI455X (gfx1250) — matrix operand order repaired and checked by reading, not compiled, not run on hardware
//
#include <hip/hip_runtime.h>
#include <hip/hip_bf16.h>

// ---------------------------------------------------------------------------
// LSTM on MI455X (gfx1250): persistent recurrent kernel with f16 WMMA.
//   prep:   f32 -> f16 weight conversion + pre-swizzle into WMMA B-fragment
//           stream; zero h state and the global barrier counter.
//   recur:  16 persistent WGs x 256 thr; per step: stage [x_t,h] (K=768, three
//           256-K chunks) in LDS as f16, GEMM via v_wmma_f32_16x16x32_f16,
//           gates+cell update with c kept in registers, device-wide split
//           barrier via monotonic atomic counter.
//   final:  h_final @ W_out + b_out through WMMA.
// ---------------------------------------------------------------------------

typedef __attribute__((ext_vector_type(16))) _Float16 v16h;
typedef __attribute__((ext_vector_type(8)))  float    v8f;
typedef __attribute__((ext_vector_type(4)))  unsigned uint4v;

union V16H { v16h v; uint4v q[2]; };

#define NWG 16

__device__ __forceinline__ float fast_sigmoid(float x) {
    return 1.0f / (1.0f + __expf(-x));
}

// ---------------------------------------------------------------------------
// Prep: swizzle gate weights [768,2048] and W_out [512,256] into per-tile
// lane/half order for the WMMA B operand (32x16 f16 tile = 512 halves:
// element (k,n) at lane = 16*(k>=16)+n, half = k%16). Also zero h and barrier.
// Workspace layout (bytes):
//   0        : Wswz     128 Ntiles * 24 Ktiles * 512 halves  (3,145,728 B)
//   3145728  : WoutSwz   16 Ntiles * 32 Ktiles * 512 halves  (  524,288 B)
//   3670016  : hstate    64*512 f16                          (   65,536 B)
//   3735552  : barrier counter (unsigned)
// ---------------------------------------------------------------------------
#define GATE_E (128 * 24 * 512)
#define WOUT_E (16 * 32 * 512)
#define HST_E  (64 * 512)

__global__ __launch_bounds__(256) void lstm_prep_kernel(
    const float* __restrict__ Wf, const float* __restrict__ Wi,
    const float* __restrict__ Wc, const float* __restrict__ Wo,
    const float* __restrict__ Wout,
    _Float16* __restrict__ Wswz, _Float16* __restrict__ WoutSwz,
    _Float16* __restrict__ hstate, unsigned* __restrict__ cnt)
{
    int idx = blockIdx.x * 256 + threadIdx.x;
    if (idx < GATE_E) {
        int tile   = idx >> 9;          // nt*24 + kt
        int within = idx & 511;
        int lane   = within >> 4;
        int h      = within & 15;
        int nt = tile / 24, kt = tile % 24;
        int klocal = ((lane >> 4) << 4) + h;     // (lane>=16)*16 + k%16
        int nl     = lane & 15;
        int n = nt * 16 + nl;                    // 0..2047 gate column
        int k = kt * 32 + klocal;                // 0..767  concat row
        int gsel = n >> 9;
        int j    = n & 511;
        const float* Wg = (gsel == 0) ? Wf : (gsel == 1) ? Wi : (gsel == 2) ? Wc : Wo;
        Wswz[idx] = (_Float16)Wg[k * 512 + j];
    } else if (idx < GATE_E + WOUT_E) {
        int i2     = idx - GATE_E;
        int tile   = i2 >> 9;           // nt*32 + kt
        int within = i2 & 511;
        int lane   = within >> 4;
        int h      = within & 15;
        int nt = tile / 32, kt = tile % 32;
        int klocal = ((lane >> 4) << 4) + h;
        int nl     = lane & 15;
        int n = nt * 16 + nl;                    // 0..255
        int k = kt * 32 + klocal;                // 0..1023
        WoutSwz[i2] = (_Float16)Wout[k * 256 + n];
    } else if (idx < GATE_E + WOUT_E + HST_E) {
        hstate[idx - GATE_E - WOUT_E] = (_Float16)0.0f;
    } else if (idx == GATE_E + WOUT_E + HST_E) {
        *cnt = 0u;
    }
}

// ---------------------------------------------------------------------------
// Persistent recurrent kernel.
// ---------------------------------------------------------------------------
__global__ __launch_bounds__(256) void lstm_recur_kernel(
    const float* __restrict__ x,                 // [64,1024,256]
    const float* __restrict__ b_f, const float* __restrict__ b_i,
    const float* __restrict__ b_c, const float* __restrict__ b_o,
    const _Float16* __restrict__ Wswz,           // pre-swizzled B fragments
    _Float16* __restrict__ hstate,               // [64,512] f16, shared state
    unsigned* __restrict__ cnt)                  // global barrier counter
{
    __shared__ __align__(16) unsigned char smem[32768];
    _Float16* chunk = reinterpret_cast<_Float16*>(smem);  // [64][256] f16
    float*    gbuf  = reinterpret_cast<float*>(smem);     // [64][128] f32 (union)

    const int tid  = threadIdx.x;
    const int wg   = blockIdx.x;     // owns h columns [wg*32, wg*32+32)
    const int lane = tid & 31;
    const int wv   = tid >> 5;       // wave 0..7
    const int g    = wv >> 1;        // gate 0..3
    const int jt   = wv & 1;         // which 16-col tile inside the slice
    const int nt_global = g * 32 + wg * 2 + jt;  // global gate N-tile 0..127
    const int grp  = lane >> 4;
    const int nl   = lane & 15;

    const int   hcol = wg * 32 + lane;           // this thread's h column
    const float bfv = b_f[hcol];
    const float biv = b_i[hcol];
    const float bcv = b_c[hcol];
    const float bov = b_o[hcol];

    float c_reg[8];                              // cell state: (m = wv + r*8, hcol)
    #pragma unroll
    for (int r = 0; r < 8; ++r) c_reg[r] = 0.0f;

    for (int t = 0; t < 1024; ++t) {
        v8f acc[4];
        #pragma unroll
        for (int mt = 0; mt < 4; ++mt)
            #pragma unroll
            for (int v = 0; v < 8; ++v) acc[mt][v] = 0.0f;

        for (int kc = 0; kc < 3; ++kc) {
            __syncthreads();                     // chunk/gbuf union safe to reuse
            if (kc == 0) {
                // K 0..255 = x_t, f32 -> f16 (coalesced: consecutive tid -> i)
                const float* xs = x + t * 256;
                #pragma unroll 4
                for (int n = 0; n < 64; ++n) {
                    int e = tid + n * 256;
                    int b = e >> 8, i = e & 255;
                    chunk[e] = (_Float16)xs[b * (1024 * 256) + i];
                }
            } else {
                // K 256..767 = h, already f16
                const _Float16* src = hstate + (kc - 1) * 256;
                #pragma unroll 4
                for (int n = 0; n < 64; ++n) {
                    int e = tid + n * 256;
                    int b = e >> 8, i = e & 255;
                    chunk[e] = src[b * 512 + i];
                }
            }
            __syncthreads();

            const _Float16* wbase =
                Wswz + (size_t)(nt_global * 24 + kc * 8) * 512 + lane * 16;
            #pragma unroll
            for (int ki = 0; ki < 8; ++ki) {
                V16H bf_;                        // B fragment: raw 32B/lane
                bf_.q[0] = *reinterpret_cast<const uint4v*>(wbase + ki * 512);
                bf_.q[1] = *reinterpret_cast<const uint4v*>(wbase + ki * 512 + 8);
                #pragma unroll
                for (int mt = 0; mt < 4; ++mt) {
                    // A 16x32 f16 layout: lane%16 = M, one 16-half K run/lane (k = 16*grp + s)
                    const int row = mt * 16 + nl;
                    const _Float16* abase = chunk + row * 256 + ki * 32 + grp * 16;
                    V16H af_;
                    af_.q[0] = *reinterpret_cast<const uint4v*>(abase);
                    af_.q[1] = *reinterpret_cast<const uint4v*>(abase + 8);
                    acc[mt] = __builtin_amdgcn_wmma_f32_16x16x32_f16(
                        false, af_.v, false, bf_.v, (short)0, acc[mt], false, false);
                }
            }
        }
        __syncthreads();
        // Scatter C tiles into the gate buffer: (m,n) -> gbuf[m][wv*16+nl]
        #pragma unroll
        for (int mt = 0; mt < 4; ++mt) {
            int cl = wv * 16 + nl;
            #pragma unroll
            for (int v = 0; v < 8; ++v) {
                int m = mt * 16 + grp * 8 + v;   // C layout: M = 8*grp + vgpr
                gbuf[m * 128 + cl] = acc[mt][v];
            }
        }
        __syncthreads();
        // Gates + cell update; this thread owns (m = wv + r*8, column hcol)
        #pragma unroll
        for (int r = 0; r < 8; ++r) {
            int m = wv + r * 8;
            float fg = fast_sigmoid(gbuf[m * 128 +  0 + lane] + bfv);
            float ig = fast_sigmoid(gbuf[m * 128 + 32 + lane] + biv);
            float ug = tanhf      (gbuf[m * 128 + 64 + lane] + bcv);
            float og = fast_sigmoid(gbuf[m * 128 + 96 + lane] + bov);
            float cn = fg * c_reg[r] + ig * ug;
            c_reg[r] = cn;
            hstate[m * 512 + hcol] = (_Float16)(og * tanhf(cn));
        }
        // Device-wide split barrier (monotonic counter): release writes,
        // acquire before next step's h reads. Only thread 0 spins.
        __threadfence();
        __syncthreads();
        if (tid == 0) {
            __hip_atomic_fetch_add(cnt, 1u, __ATOMIC_RELEASE, __HIP_MEMORY_SCOPE_AGENT);
            const unsigned target = (unsigned)(t + 1) * NWG;
            while (__hip_atomic_load(cnt, __ATOMIC_ACQUIRE, __HIP_MEMORY_SCOPE_AGENT) < target) {
                __builtin_amdgcn_s_sleep(1);
            }
        }
        __syncthreads();
        __threadfence();
    }
}

// ---------------------------------------------------------------------------
// Epilogue: out = h_final @ W_out + b_out   ([64,512] @ [512,256])
// ---------------------------------------------------------------------------
__global__ __launch_bounds__(256) void lstm_final_kernel(
    const _Float16* __restrict__ hstate,
    const _Float16* __restrict__ WoutSwz,
    const float* __restrict__ b_out,
    float* __restrict__ out)
{
    __shared__ __align__(16) _Float16 hsm[64 * 512];   // 64 KB
    const int tid  = threadIdx.x;
    const int lane = tid & 31;
    const int wv   = tid >> 5;
    const int grp  = lane >> 4;
    const int nl   = lane & 15;

    const unsigned* hs32 = reinterpret_cast<const unsigned*>(hstate);
    unsigned*       sm32 = reinterpret_cast<unsigned*>(hsm);
    #pragma unroll 4
    for (int n = 0; n < 64; ++n) sm32[tid + n * 256] = hs32[tid + n * 256];
    __syncthreads();

    for (int p = 0; p < 2; ++p) {
        const int nt = wv * 2 + p;                     // N-tile 0..15
        v8f acc[4];
        #pragma unroll
        for (int mt = 0; mt < 4; ++mt)
            #pragma unroll
            for (int v = 0; v < 8; ++v) acc[mt][v] = 0.0f;

        const _Float16* wbase = WoutSwz + (size_t)(nt * 32) * 512 + lane * 16;
        for (int kt = 0; kt < 32; ++kt) {
            V16H bf_;
            bf_.q[0] = *reinterpret_cast<const uint4v*>(wbase + kt * 512);
            bf_.q[1] = *reinterpret_cast<const uint4v*>(wbase + kt * 512 + 8);
            #pragma unroll
            for (int mt = 0; mt < 4; ++mt) {
                const int row = mt * 16 + nl;
                const _Float16* abase = hsm + row * 512 + kt * 32 + grp * 16;
                V16H af_;
                af_.q[0] = *reinterpret_cast<const uint4v*>(abase);
                af_.q[1] = *reinterpret_cast<const uint4v*>(abase + 8);
                acc[mt] = __builtin_amdgcn_wmma_f32_16x16x32_f16(
                    false, af_.v, false, bf_.v, (short)0, acc[mt], false, false);
            }
        }
        const int n  = nt * 16 + nl;
        const float bo = b_out[n];
        #pragma unroll
        for (int mt = 0; mt < 4; ++mt)
            #pragma unroll
            for (int v = 0; v < 8; ++v) {
                int m = mt * 16 + grp * 8 + v;
                out[m * 256 + n] = acc[mt][v] + bo;
            }
    }
}

// ---------------------------------------------------------------------------
extern "C" void kernel_launch(void* const* d_in, const int* in_sizes, int n_in,
                              void* d_out, int out_size, void* d_ws, size_t ws_size,
                              hipStream_t stream) {
    const float* x    = (const float*)d_in[0];
    const float* Wf   = (const float*)d_in[1];
    const float* bf   = (const float*)d_in[2];
    const float* Wi   = (const float*)d_in[3];
    const float* bi   = (const float*)d_in[4];
    const float* Wc   = (const float*)d_in[5];
    const float* bc   = (const float*)d_in[6];
    const float* Wo   = (const float*)d_in[7];
    const float* bo   = (const float*)d_in[8];
    const float* Wout = (const float*)d_in[9];
    const float* bout = (const float*)d_in[10];

    char* ws = (char*)d_ws;
    _Float16* Wswz    = (_Float16*)(ws);
    _Float16* WoutSwz = (_Float16*)(ws + 3145728);
    _Float16* hstate  = (_Float16*)(ws + 3670016);
    unsigned* cnt     = (unsigned*)(ws + 3735552);

    const int prep_elems = GATE_E + WOUT_E + HST_E + 1;
    const int prep_grid  = (prep_elems + 255) / 256;

    lstm_prep_kernel<<<prep_grid, 256, 0, stream>>>(
        Wf, Wi, Wc, Wo, Wout, Wswz, WoutSwz, hstate, cnt);
    lstm_recur_kernel<<<NWG, 256, 0, stream>>>(
        x, bf, bi, bc, bo, Wswz, hstate, cnt);
    lstm_final_kernel<<<1, 256, 0, stream>>>(
        hstate, WoutSwz, bout, (float*)d_out);
}
